// GATNet_69028714381979
// MI455X (gfx1250) — matrix operand order repaired and checked by reading, not compiled, not run on hardware
//
#include <hip/hip_runtime.h>
#include <math.h>

#define NN      50000
#define DIMC    128
#define HEADS   8
#define HD      16
#define LLAY    2
#define EE      800000
#define NRELC   2000
#define TTR     200000
#define RRR     50000
#define BBB     10000
#define ALPHA   0.2f
#define MTILES  5          // row-tiles per block (50000 = 625 * 80)

typedef __attribute__((ext_vector_type(16))) _Float16 v16h;
typedef __attribute__((ext_vector_type(8)))  _Float16 v8h;
typedef __attribute__((ext_vector_type(8)))  float    v8f;

// ---------------- elementwise f32 -> f16 ----------------
__global__ void k_cvt_f16(const float* __restrict__ in, _Float16* __restrict__ out, int n) {
    int i = blockIdx.x * blockDim.x + threadIdx.x;
    if (i < n) out[i] = (_Float16)in[i];
}

// ---------------- W f32 row-major -> f16 column-major (transpose at convert) ----------------
__global__ void k_cvt_wT_f16(const float* __restrict__ in, _Float16* __restrict__ out) {
    int i = blockIdx.x * blockDim.x + threadIdx.x;   // i = n*DIMC + k
    if (i >= DIMC * DIMC) return;
    int n = i / DIMC, k = i % DIMC;
    out[i] = (_Float16)in[k * DIMC + n];
}

// ---------------- WMMA GEMM: h[N x 128] = x[N x 128] * W[128 x 128] ----------------
// Block = 256 threads = 8 waves; covers 80 rows (5 row-tiles) x all 128 cols.
// Wave w owns col-tile [16w,16w+16), loads each B fragment once per K-chunk and
// reuses it across 5 A tiles -> 20 WMMAs / wave. W staged in LDS column-major so
// B fragments are contiguous (2x ds_load_b128, no sub-word packing).
__global__ __launch_bounds__(256) void k_gemm_wmma(const _Float16* __restrict__ xh,
                                                   const _Float16* __restrict__ whT,
                                                   float* __restrict__ hout) {
    __shared__ __align__(16) _Float16 wl[DIMC * DIMC];   // [n][k] column-major
    {
        const float4* src = (const float4*)whT;
        float4*       dst = (float4*)wl;
        for (int i = threadIdx.x; i < DIMC * DIMC / 8; i += 256) dst[i] = src[i];
    }
    __syncthreads();

    const int wave  = threadIdx.x >> 5;
    const int lane  = threadIdx.x & 31;
    const int half_ = lane >> 4;
    const int r     = lane & 15;
    const int n0    = wave * 16;
    const int mbase = blockIdx.x * (16 * MTILES);

    v8f acc[MTILES] = {};
    const v8h* bcol = (const v8h*)(wl + (n0 + r) * DIMC);

#pragma unroll
    for (int kc = 0; kc < DIMC; kc += 32) {
        // B fragment: lane holds col n0+r, K = kc + 16*half_ .. +15 (contiguous)
        v8h blo = bcol[(kc >> 3) + 2 * half_];
        v8h bhi = bcol[(kc >> 3) + 2 * half_ + 1];
        v16h b = __builtin_shufflevector(blo, bhi,
                                         0, 1, 2, 3, 4, 5, 6, 7,
                                         8, 9, 10, 11, 12, 13, 14, 15);
#pragma unroll
        for (int t = 0; t < MTILES; ++t) {
            // A fragment: lane holds row mbase+16t+r, K = kc + 16*half_ .. +15 (contiguous)
            const v8h* arow8 = (const v8h*)(xh + (size_t)(mbase + t * 16 + r) * DIMC);
            v8h alo = arow8[(kc >> 3) + 2 * half_];
            v8h ahi = arow8[(kc >> 3) + 2 * half_ + 1];
            v16h a = __builtin_shufflevector(alo, ahi,
                                             0, 1, 2, 3, 4, 5, 6, 7,
                                             8, 9, 10, 11, 12, 13, 14, 15);
            acc[t] = __builtin_amdgcn_wmma_f32_16x16x32_f16(false, a, false, b,
                                                            (short)0, acc[t], false, false);
        }
    }
#pragma unroll
    for (int t = 0; t < MTILES; ++t)
#pragma unroll
        for (int v = 0; v < 8; ++v)
            hout[(size_t)(mbase + t * 16 + v + half_ * 8) * DIMC + n0 + r] = acc[t][v];
}

// ---------------- per-node attention dots ----------------
__global__ void k_att(const float* __restrict__ h, const float* __restrict__ a_s,
                      const float* __restrict__ a_d, float* __restrict__ atts,
                      float* __restrict__ attd) {
    int i = blockIdx.x * blockDim.x + threadIdx.x;   // n*HEADS + head
    if (i >= NN * HEADS) return;
    int n = i / HEADS, hd = i % HEADS;
    const float* hr = h + (size_t)n * DIMC + hd * HD;
    const float* as = a_s + hd * HD;
    const float* ad = a_d + hd * HD;
    float s = 0.f, d = 0.f;
#pragma unroll
    for (int k = 0; k < HD; ++k) { s += hr[k] * as[k]; d += hr[k] * ad[k]; }
    atts[i] = s; attd[i] = d;
}

// ---------------- init accumulators ----------------
__global__ void k_init(float* __restrict__ xacc, float* __restrict__ m, float* __restrict__ den) {
    int i = blockIdx.x * blockDim.x + threadIdx.x;
    if (i < NN * DIMC) xacc[i] = 0.f;
    if (i < NN * HEADS) { m[i] = -__builtin_inff(); den[i] = 0.f; }
}

// float atomic max via sign-split int atomics (total order preserved across views)
__device__ __forceinline__ void atomicMaxF(float* addr, float v) {
    if (v >= 0.f) atomicMax((int*)addr, __float_as_int(v));
    else          atomicMin((unsigned int*)addr, __float_as_uint(v));
}

// ---------------- edge pass 1: segment max of leaky-relu logits ----------------
__global__ void k_edge_max(const int* __restrict__ src, const int* __restrict__ dst,
                           const float* __restrict__ atts, const float* __restrict__ attd,
                           float* __restrict__ m) {
    int i = blockIdx.x * blockDim.x + threadIdx.x;   // e*HEADS + head
    if (i >= EE * HEADS) return;
    int e = i / HEADS, hd = i % HEADS;
    int s = src[e], d = dst[e];
    float v = atts[s * HEADS + hd] + attd[d * HEADS + hd];
    v = v > 0.f ? v : ALPHA * v;
    atomicMaxF(&m[d * HEADS + hd], v);
}

// ---------------- edge pass 2: softmax weights, scatter-add denom + messages ----------------
__global__ void k_edge_sum(const int* __restrict__ src, const int* __restrict__ dst,
                           const float* __restrict__ atts, const float* __restrict__ attd,
                           const float* __restrict__ m, const float* __restrict__ h,
                           float* __restrict__ den, float* __restrict__ xacc) {
    int i = blockIdx.x * blockDim.x + threadIdx.x;   // e*HEADS + head
    if (i >= EE * HEADS) return;
    int e = i / HEADS, hd = i % HEADS;
    int s = src[e], d = dst[e];
    float v = atts[s * HEADS + hd] + attd[d * HEADS + hd];
    v = v > 0.f ? v : ALPHA * v;
    float w = __expf(v - m[d * HEADS + hd]);
    atomicAdd(&den[d * HEADS + hd], w);
    const float* hs = h + (size_t)s * DIMC + hd * HD;
    float*       xa = xacc + (size_t)d * DIMC + hd * HD;
    __builtin_prefetch(hs, 0, 0);
#pragma unroll
    for (int k = 0; k < HD; ++k) atomicAdd(&xa[k], w * hs[k]);
}

// ---------------- finalize: divide by denom, optional ELU ----------------
__global__ void k_finalize(float* __restrict__ xacc, const float* __restrict__ den, int elu) {
    int i = blockIdx.x * blockDim.x + threadIdx.x;
    if (i >= NN * DIMC) return;
    int n = i / DIMC, hd = (i % DIMC) / HD;
    float v = xacc[i] / fmaxf(den[n * HEADS + hd], 1e-16f);
    if (elu) v = v > 0.f ? v : (__expf(v) - 1.f);
    xacc[i] = v;
}

// ---------------- gather g[sel] rows ----------------
__global__ void k_gather(const float* __restrict__ g, const int* __restrict__ idx,
                         float* __restrict__ out) {
    int i = blockIdx.x * blockDim.x + threadIdx.x;
    if (i >= BBB * DIMC) return;
    int b = i / DIMC, d = i % DIMC;
    out[i] = g[(size_t)idx[b] * DIMC + d];
}

// ---------------- row-wise l2 norm (one wave32 per 128-dim row) ----------------
__global__ void k_l2norm(const float* __restrict__ in, float* __restrict__ out, int rows) {
    int wid  = (blockIdx.x * blockDim.x + threadIdx.x) >> 5;
    int lane = threadIdx.x & 31;
    if (wid >= rows) return;
    const float* rp = in + (size_t)wid * DIMC;
    float s = 0.f;
#pragma unroll
    for (int k = 0; k < 4; ++k) { float v = rp[lane + 32 * k]; s += v * v; }
#pragma unroll
    for (int off = 16; off > 0; off >>= 1) s += __shfl_xor(s, off, 32);
    float sc = 1.f / fmaxf(sqrtf(s), 1e-12f);
    float* op = out + (size_t)wid * DIMC;
#pragma unroll
    for (int k = 0; k < 4; ++k) op[lane + 32 * k] = rp[lane + 32 * k] * sc;
}

// ---------------- TransE truth value (one wave per row) ----------------
__global__ void k_transe(const float* __restrict__ ng, const float* __restrict__ nrel,
                         const int* __restrict__ hi, const int* __restrict__ ti,
                         const int* __restrict__ ri_, float* __restrict__ tv, int rows) {
    int wid  = (blockIdx.x * blockDim.x + threadIdx.x) >> 5;
    int lane = threadIdx.x & 31;
    if (wid >= rows) return;
    const float* ph = ng   + (size_t)hi[wid]  * DIMC;
    const float* pt = ng   + (size_t)ti[wid]  * DIMC;
    const float* pr = nrel + (size_t)ri_[wid] * DIMC;
    float s = 0.f;
#pragma unroll
    for (int k = 0; k < 4; ++k) {
        int d = lane + 32 * k;
        s += fabsf(ph[d] + pr[d] - pt[d]);
    }
#pragma unroll
    for (int off = 16; off > 0; off >>= 1) s += __shfl_xor(s, off, 32);
    if (lane == 0) tv[wid] = 1.f - s * (1.f / (3.f * 11.313708498984761f));
}

// ---------------- rule truth value (one wave per rule) ----------------
__global__ void k_rule(const float* __restrict__ ng, const float* __restrict__ nrel,
                       const int* __restrict__ rh, const int* __restrict__ rt,
                       const int* __restrict__ rrel, const int* __restrict__ prem,
                       const float* __restrict__ tv, int tv_stride, int tv_off,
                       float* __restrict__ out) {
    int wid  = (blockIdx.x * blockDim.x + threadIdx.x) >> 5;
    int lane = threadIdx.x & 31;
    if (wid >= RRR) return;
    const float* ph = ng   + (size_t)rh[wid]   * DIMC;
    const float* pt = ng   + (size_t)rt[wid]   * DIMC;
    const float* pr = nrel + (size_t)rrel[wid] * DIMC;
    float s = 0.f;
#pragma unroll
    for (int k = 0; k < 4; ++k) {
        int d = lane + 32 * k;
        s += fabsf(ph[d] + pr[d] - pt[d]);
    }
#pragma unroll
    for (int off = 16; off > 0; off >>= 1) s += __shfl_xor(s, off, 32);
    if (lane == 0) {
        float rs = 1.f - s * (1.f / (3.f * 11.313708498984761f));
        int p0 = prem[wid * 2 + 0], p1 = prem[wid * 2 + 1];
        float f1 = (p0 < TTR) ? tv[(size_t)p0 * tv_stride + tv_off] : 1.f;
        float f2 = (p1 < TTR) ? tv[(size_t)p1 * tv_stride + tv_off] : 1.f;
        out[wid] = 1.f + f1 * f2 * (rs - 1.f);
    }
}

extern "C" void kernel_launch(void* const* d_in, const int* in_sizes, int n_in,
                              void* d_out, int out_size, void* d_ws, size_t ws_size,
                              hipStream_t stream) {
    (void)in_sizes; (void)n_in; (void)out_size; (void)ws_size;
    const float* ent_sr  = (const float*)d_in[0];
    const float* ent_tg  = (const float*)d_in[1];
    const float* rel_sr  = (const float*)d_in[2];
    const float* rel_tg  = (const float*)d_in[3];
    const float* gat_W   = (const float*)d_in[4];
    const float* a_src   = (const float*)d_in[5];
    const float* a_dst   = (const float*)d_in[6];
    const int*   sr_data = (const int*)d_in[7];
    const int*   tg_data = (const int*)d_in[8];
    const int*   edges_sr = (const int*)d_in[9];
    const int*   edges_tg = (const int*)d_in[10];
    const int*   h_sr = (const int*)d_in[11];
    const int*   t_sr = (const int*)d_in[12];
    const int*   r_sr = (const int*)d_in[13];
    const int*   h_tg = (const int*)d_in[14];
    const int*   t_tg = (const int*)d_in[15];
    const int*   r_tg = (const int*)d_in[16];
    const int*   rh_sr = (const int*)d_in[17];
    const int*   rt_sr = (const int*)d_in[18];
    const int*   rr_sr = (const int*)d_in[19];
    const int*   prem_sr = (const int*)d_in[20];
    const int*   rh_tg = (const int*)d_in[21];
    const int*   rt_tg = (const int*)d_in[22];
    const int*   rr_tg = (const int*)d_in[23];
    const int*   prem_tg = (const int*)d_in[24];

    char* ws = (char*)d_ws;
    size_t off = 0;
    auto alloc = [&](size_t bytes) -> void* {
        void* p = ws + off;
        off = (off + bytes + 255) & ~(size_t)255;
        return p;
    };
    float*    xacc      = (float*)alloc((size_t)NN * DIMC * 4);
    float*    hbuf      = (float*)alloc((size_t)NN * DIMC * 4);
    _Float16* xf16      = (_Float16*)alloc((size_t)NN * DIMC * 2);
    _Float16* wf16T     = (_Float16*)alloc((size_t)DIMC * DIMC * 2);
    float*    atts      = (float*)alloc((size_t)NN * HEADS * 4);
    float*    attd      = (float*)alloc((size_t)NN * HEADS * 4);
    float*    mbuf      = (float*)alloc((size_t)NN * HEADS * 4);
    float*    den       = (float*)alloc((size_t)NN * HEADS * 4);
    float*    ng_sr     = (float*)alloc((size_t)NN * DIMC * 4);
    float*    ng_tg     = (float*)alloc((size_t)NN * DIMC * 4);
    float*    nrel_sr_b = (float*)alloc((size_t)NRELC * DIMC * 4);
    float*    nrel_tg_b = (float*)alloc((size_t)NRELC * DIMC * 4);

    float* out         = (float*)d_out;
    float* out_gsr     = out;                                  // B*128
    float* out_gtg     = out + (size_t)BBB * DIMC;             // B*128
    float* out_tv_sr   = out + (size_t)2 * BBB * DIMC;         // T*2
    float* out_tv_tg   = out_tv_sr + (size_t)TTR * 2;          // T*2
    float* out_rule_sr = out_tv_tg + (size_t)TTR * 2;          // R
    float* out_rule_tg = out_rule_sr + RRR;                    // R

    const int TB = 256;
    for (int graph = 0; graph < 2; ++graph) {
        const float* ent  = graph ? ent_tg : ent_sr;
        const int* edges  = graph ? edges_tg : edges_sr;
        const int* esrc   = edges;        // edges[0] row
        const int* edst   = edges + EE;   // edges[1] row
        for (int l = 0; l < LLAY; ++l) {
            const float* xin = (l == 0) ? ent : xacc;
            k_cvt_f16<<<(NN * DIMC + TB - 1) / TB, TB, 0, stream>>>(xin, xf16, NN * DIMC);
            k_cvt_wT_f16<<<(DIMC * DIMC + TB - 1) / TB, TB, 0, stream>>>(
                gat_W + (size_t)l * DIMC * DIMC, wf16T);
            k_gemm_wmma<<<NN / (16 * MTILES), 256, 0, stream>>>(xf16, wf16T, hbuf);
            k_att<<<(NN * HEADS + TB - 1) / TB, TB, 0, stream>>>(
                hbuf, a_src + l * HEADS * HD, a_dst + l * HEADS * HD, atts, attd);
            k_init<<<(NN * DIMC + TB - 1) / TB, TB, 0, stream>>>(xacc, mbuf, den);
            k_edge_max<<<(EE * HEADS + TB - 1) / TB, TB, 0, stream>>>(esrc, edst, atts, attd, mbuf);
            k_edge_sum<<<(EE * HEADS + TB - 1) / TB, TB, 0, stream>>>(
                esrc, edst, atts, attd, mbuf, hbuf, den, xacc);
            k_finalize<<<(NN * DIMC + TB - 1) / TB, TB, 0, stream>>>(
                xacc, den, (l < LLAY - 1) ? 1 : 0);
        }
        const int* sel = graph ? tg_data : sr_data;
        float* og = graph ? out_gtg : out_gsr;
        k_gather<<<(BBB * DIMC + TB - 1) / TB, TB, 0, stream>>>(xacc, sel, og);
        float* ng = graph ? ng_tg : ng_sr;
        k_l2norm<<<(NN + 7) / 8, 256, 0, stream>>>(xacc, ng, NN);
    }
    k_l2norm<<<(NRELC + 7) / 8, 256, 0, stream>>>(rel_sr, nrel_sr_b, NRELC);
    k_l2norm<<<(NRELC + 7) / 8, 256, 0, stream>>>(rel_tg, nrel_tg_b, NRELC);

    const int trows = TTR * 2;  // (T,2) flattened
    k_transe<<<(trows + 7) / 8, 256, 0, stream>>>(ng_sr, nrel_sr_b, h_sr, t_sr, r_sr, out_tv_sr, trows);
    k_transe<<<(trows + 7) / 8, 256, 0, stream>>>(ng_tg, nrel_tg_b, h_tg, t_tg, r_tg, out_tv_tg, trows);

    // rules: sr uses sr_tv[:,0] (stride 2, offset 0); tg uses tg_tv[:,1] (stride 2, offset 1)
    k_rule<<<(RRR + 7) / 8, 256, 0, stream>>>(ng_sr, nrel_sr_b, rh_sr, rt_sr, rr_sr, prem_sr,
                                              out_tv_sr, 2, 0, out_rule_sr);
    k_rule<<<(RRR + 7) / 8, 256, 0, stream>>>(ng_tg, nrel_tg_b, rh_tg, rt_tg, rr_tg, prem_tg,
                                              out_tv_tg, 2, 1, out_rule_tg);
}
